// Bottleneck_57561151701110
// MI455X (gfx1250) — matrix operand order repaired and checked by reading, not compiled, not run on hardware
//
#include <hip/hip_runtime.h>
#include <hip/hip_bf16.h>

// ---------------- problem constants (match reference setup_inputs) ----------
#define BATCH 8
#define CIN   256
#define Hh    56
#define Ww    56
#define SPAT  (Hh * Ww)         // 3136 = 196 * 16 = 98 * 32
#define REL   32
#define MID   256
#define OUTC  256
#define K2    9
#define SHARE 8
#define GRP   (MID / SHARE)     // 32
#define YCH   (REL + REL + MID) // 320 fused channels: x1 | x2 | x3
#define NEG_SLOPE 0.01f
#define BN_EPS    1e-5f

// ---------------- WMMA vector types ----------------------------------------
typedef __attribute__((ext_vector_type(16))) __bf16 v16bf;
typedef __attribute__((ext_vector_type(8)))  float  v8f;

union FragBF16 { v16bf v; unsigned short u[16]; uint4 q[2]; };
union Pack8    { uint4 q; unsigned short u[8]; };

// round-to-nearest-even f32 -> bf16 bits
__device__ __forceinline__ unsigned short f2bf(float f) {
    union { float f; unsigned int u; } c; c.f = f;
    unsigned int u = c.u;
    unsigned int r = (u + 0x7FFFu + ((u >> 16) & 1u)) >> 16;
    return (unsigned short)r;
}

__device__ __forceinline__ float leaky(float v) {
    return v > 0.f ? v : NEG_SLOPE * v;
}

// ============================================================================
// Prep: pack fused [conv1|conv2|conv3] weights (320x256) + fused bias (320)
//       and convo weights (256x256) into bf16 — removes all per-lane
//       pointer/bias selection from the GEMM hot paths.
// ============================================================================
__global__ __launch_bounds__(256)
void k_prep(const float* __restrict__ w1c, const float* __restrict__ b1c,
            const float* __restrict__ w2c, const float* __restrict__ b2c,
            const float* __restrict__ w3c, const float* __restrict__ b3c,
            const float* __restrict__ wo,
            unsigned short* __restrict__ Wf, float* __restrict__ biasf,
            unsigned short* __restrict__ Wo) {
    const int t = blockIdx.x * blockDim.x + threadIdx.x;
    const int NW1 = YCH * CIN;    // 81920
    const int NW2 = OUTC * MID;   // 65536
    if (t < NW1) {
        const int row = t / CIN, k = t % CIN;
        const float v = (row < REL)     ? w1c[row * CIN + k]
                      : (row < 2 * REL) ? w2c[(row - REL) * CIN + k]
                                        : w3c[(row - 2 * REL) * CIN + k];
        Wf[t] = f2bf(v);
    } else if (t < NW1 + NW2) {
        const int u = t - NW1;
        Wo[u] = f2bf(wo[u]);
    } else if (t < NW1 + NW2 + YCH) {
        const int o = t - NW1 - NW2;
        biasf[o] = (o < REL) ? b1c[o] : (o < 2 * REL) ? b2c[o - REL] : b3c[o - 2 * REL];
    }
}

// ============================================================================
// Transpose + downconvert: x (B,C,S) f32  ->  xT (B,S,C) bf16
//   LDS-tiled 32x32 so both global read and write are coalesced.
// ============================================================================
__global__ __launch_bounds__(256)
void k_transpose_x(const float* __restrict__ x, unsigned short* __restrict__ xT) {
    __shared__ unsigned short tile[32][34];   // +2 pad: avoid bank conflicts
    const int ST = SPAT / 32;                 // 98
    const int CT = CIN / 32;                  // 8
    int blk = blockIdx.x;
    const int b  = blk / (ST * CT);
    const int r  = blk % (ST * CT);
    const int ct = r / ST;
    const int st = r % ST;
    const int tx = threadIdx.x & 31;
    const int ty = threadIdx.x >> 5;          // 0..7

    const float* xb = x + (size_t)b * CIN * SPAT;
    #pragma unroll
    for (int i = 0; i < 32; i += 8) {
        const int c = ct * 32 + ty + i;
        const int s = st * 32 + tx;
        tile[ty + i][tx] = f2bf(xb[(size_t)c * SPAT + s]);
    }
    __syncthreads();
    unsigned short* ob = xT + (size_t)b * SPAT * CIN;
    #pragma unroll
    for (int i = 0; i < 32; i += 8) {
        const int s = st * 32 + ty + i;
        const int c = ct * 32 + tx;
        ob[(size_t)s * CIN + c] = tile[tx][ty + i];
    }
}

// ============================================================================
// Kernel 1: fused 1x1 conv GEMM  Y[b, 0:320, s] = [conv1|conv2|conv3](x)
//   one wave32 per 16x16 tile; A/B fragments are 2x b128 contiguous loads.
// ============================================================================
__global__ __launch_bounds__(256)
void k_qkv_gemm(const unsigned short* __restrict__ Wf,
                const float* __restrict__ biasf,
                const unsigned short* __restrict__ xT,
                float* __restrict__ Y) {
    const int NT = SPAT / 16;                 // 196
    const int MT = YCH / 16;                  // 20
    const int wave = blockIdx.x * (blockDim.x >> 5) + (threadIdx.x >> 5);
    const int lane = threadIdx.x & 31;
    const int tilesPerB = NT * MT;            // 3920
    const int b  = wave / tilesPerB;
    if (b >= BATCH) return;                   // wave-uniform (grid is exact)
    const int rr = wave % tilesPerB;
    const int mt = rr / NT;
    const int nt = rr % NT;

    const int lg = lane >> 4;
    const int lm = lane & 15;
    const int row  = mt * 16 + lm;            // A row for this lane
    const int scol = nt * 16 + lm;            // B column for this lane

    const unsigned short* wrow = Wf + (size_t)row * CIN;
    const unsigned short* xrow = xT + ((size_t)b * SPAT + scol) * CIN;

    v8f acc;
    #pragma unroll
    for (int i = 0; i < 8; ++i) acc[i] = 0.f;

    #pragma unroll
    for (int kk = 0; kk < CIN; kk += 32) {
        FragBF16 A, B;
        // A 16x32 bf16 layout: lane holds k = (e&7) + 8*lg + 16*(e>>3)
        A.q[0] = *(const uint4*)(wrow + kk + 8 * lg);
        A.q[1] = *(const uint4*)(wrow + kk + 16 + 8 * lg);
        // B 32x16 bf16 layout: lane holds k = (e&7) + 8*lg + 16*(e>>3)
        B.q[0] = *(const uint4*)(xrow + kk + 8 * lg);
        B.q[1] = *(const uint4*)(xrow + kk + 16 + 8 * lg);
        acc = __builtin_amdgcn_wmma_f32_16x16x32_bf16(
                  false, A.v, false, B.v, (short)0, acc, false, false);
    }

    float* yb = Y + (size_t)b * YCH * SPAT;
    #pragma unroll
    for (int i = 0; i < 8; ++i) {
        const int o = mt * 16 + i + 8 * lg;   // D layout: VGPR i -> M = i + 8*lg
        yb[(size_t)o * SPAT + scol] = acc[i] + biasf[o];
    }
}

// ============================================================================
// Kernel 2: attention logits per (b, j, s):
//   feat = [x1 - unfold(x2), p - unfold(p)] ; BN+ReLU ; w1 ; BN+ReLU ; w2+b
// ============================================================================
__global__ __launch_bounds__(256)
void k_logits(const float* __restrict__ Y,
              const float* __restrict__ cpw, const float* __restrict__ cpb,
              const float* __restrict__ bag, const float* __restrict__ bab,
              const float* __restrict__ bam, const float* __restrict__ bav,
              const float* __restrict__ w1,
              const float* __restrict__ bbg, const float* __restrict__ bbb,
              const float* __restrict__ bbm, const float* __restrict__ bbv,
              const float* __restrict__ w2,  const float* __restrict__ w2b,
              float* __restrict__ att) {
    const int t = blockIdx.x * blockDim.x + threadIdx.x;
    if (t >= BATCH * K2 * SPAT) return;
    const int s   = t % SPAT;
    const int rem = t / SPAT;
    const int j   = rem % K2;
    const int b   = rem / K2;

    const int yy = s / Ww, xx = s % Ww;
    const int yn = yy + j / 3 - 1;
    const int xn = xx + j % 3 - 1;
    const bool inb = (yn >= 0) && (yn < Hh) && (xn >= 0) && (xn < Ww);
    const int sn = inb ? (yn * Ww + xn) : 0;

    const float* yb = Y + (size_t)b * YCH * SPAT;

    float f[REL + 2];
    #pragma unroll
    for (int c = 0; c < REL; ++c) {
        const float a = yb[(size_t)c * SPAT + s];                      // x1 center
        const float v = inb ? yb[(size_t)(REL + c) * SPAT + sn] : 0.f; // x2 nbr
        f[c] = a - v;
    }
    // position channels (zero padding outside -> p_center - 0)
    const float lw  = -1.f + 2.f * (float)xx / (float)(Ww - 1);
    const float lh  = -1.f + 2.f * (float)yy / (float)(Hh - 1);
    const float lwn = -1.f + 2.f * (float)xn / (float)(Ww - 1);
    const float lhn = -1.f + 2.f * (float)yn / (float)(Hh - 1);
    #pragma unroll
    for (int c = 0; c < 2; ++c) {
        const float pc = cpw[c * 2 + 0] * lw  + cpw[c * 2 + 1] * lh  + cpb[c];
        const float pn = inb ? (cpw[c * 2 + 0] * lwn + cpw[c * 2 + 1] * lhn + cpb[c]) : 0.f;
        f[REL + c] = pc - pn;
    }
    // BN-a + ReLU
    #pragma unroll
    for (int c = 0; c < REL + 2; ++c) {
        const float inv = bag[c] * rsqrtf(bav[c] + BN_EPS);
        const float v = f[c] * inv + (bab[c] - bam[c] * inv);
        f[c] = v > 0.f ? v : 0.f;
    }
    // w1 (32x34, no bias) then BN-b + ReLU
    float h[REL];
    for (int o = 0; o < REL; ++o) {
        float a = 0.f;
        #pragma unroll
        for (int c = 0; c < REL + 2; ++c) a = fmaf(w1[o * (REL + 2) + c], f[c], a);
        const float inv = bbg[o] * rsqrtf(bbv[o] + BN_EPS);
        const float v = a * inv + (bbb[o] - bbm[o] * inv);
        h[o] = v > 0.f ? v : 0.f;
    }
    // w2 (32x32) + bias -> logits, layout att[b][g][j][s]
    float* ab = att + (size_t)b * GRP * K2 * SPAT;
    for (int o = 0; o < GRP; ++o) {
        float a = w2b[o];
        #pragma unroll
        for (int c = 0; c < REL; ++c) a = fmaf(w2[o * REL + c], h[c], a);
        ab[(size_t)(o * K2 + j) * SPAT + s] = a;
    }
}

// ============================================================================
// Kernel 3: softmax over the 9-neighbor axis (in place)
// ============================================================================
__global__ __launch_bounds__(256)
void k_softmax(float* __restrict__ att) {
    const int t = blockIdx.x * blockDim.x + threadIdx.x;
    if (t >= BATCH * GRP * SPAT) return;
    const int s   = t % SPAT;
    const int rem = t / SPAT;
    const int g   = rem % GRP;
    const int b   = rem / GRP;
    float* p = att + ((size_t)(b * GRP + g) * K2) * SPAT + s;

    float v[K2];
    float m = -3.0e38f;
    #pragma unroll
    for (int j = 0; j < K2; ++j) { v[j] = p[(size_t)j * SPAT]; m = fmaxf(m, v[j]); }
    float sum = 0.f;
    #pragma unroll
    for (int j = 0; j < K2; ++j) { v[j] = __expf(v[j] - m); sum += v[j]; }
    const float inv = 1.f / sum;
    #pragma unroll
    for (int j = 0; j < K2; ++j) p[(size_t)j * SPAT] = v[j] * inv;
}

// ============================================================================
// Kernel 4: grouped aggregation of unfolded x3 + LeakyReLU.
//   One thread per (b, group, s): the 8 channels of a share-group use the SAME
//   attention weights; writes 8 bf16 (one 16B store) straight into the
//   transposed activation layout actT[b][s][c] consumed by the convo GEMM.
// ============================================================================
__global__ __launch_bounds__(256)
void k_aggregate(const float* __restrict__ Y, const float* __restrict__ att,
                 unsigned short* __restrict__ actT) {
    const int t = blockIdx.x * blockDim.x + threadIdx.x;
    if (t >= BATCH * GRP * SPAT) return;
    const int s   = t % SPAT;
    const int rem = t / SPAT;
    const int g   = rem % GRP;
    const int b   = rem / GRP;
    const int yy = s / Ww, xx = s % Ww;

    const float* x3 = Y   + (size_t)(b * YCH + 2 * REL + g * SHARE) * SPAT;
    const float* wp = att + ((size_t)(b * GRP + g) * K2) * SPAT + s;

    float acc[SHARE];
    #pragma unroll
    for (int c = 0; c < SHARE; ++c) acc[c] = 0.f;

    #pragma unroll
    for (int j = 0; j < K2; ++j) {
        const int yn = yy + j / 3 - 1;
        const int xn = xx + j % 3 - 1;
        const bool inb = (yn >= 0) && (yn < Hh) && (xn >= 0) && (xn < Ww);
        const int sn = yn * Ww + xn;
        const float wj = wp[(size_t)j * SPAT];
        #pragma unroll
        for (int c = 0; c < SHARE; ++c) {
            const float xv = inb ? x3[(size_t)c * SPAT + sn] : 0.f;
            acc[c] = fmaf(xv, wj, acc[c]);
        }
    }
    Pack8 p;
    #pragma unroll
    for (int c = 0; c < SHARE; ++c) p.u[c] = f2bf(leaky(acc[c]));
    *(uint4*)(actT + ((size_t)(b * SPAT + s) * MID + g * SHARE)) = p.q;
}

// ============================================================================
// Kernel 5: convo GEMM (bf16 WMMA) + bias + LeakyReLU + residual -> d_out
// ============================================================================
__global__ __launch_bounds__(256)
void k_out_gemm(const unsigned short* __restrict__ actT,
                const unsigned short* __restrict__ Wo,
                const float* __restrict__ bo,
                const float* __restrict__ x, float* __restrict__ out) {
    const int NT = SPAT / 16;                 // 196
    const int MT = OUTC / 16;                 // 16
    const int wave = blockIdx.x * (blockDim.x >> 5) + (threadIdx.x >> 5);
    const int lane = threadIdx.x & 31;
    const int tilesPerB = NT * MT;            // 3136
    const int b  = wave / tilesPerB;
    if (b >= BATCH) return;
    const int rr = wave % tilesPerB;
    const int mt = rr / NT;
    const int nt = rr % NT;

    const int lg = lane >> 4;
    const int lm = lane & 15;
    const int row  = mt * 16 + lm;
    const int scol = nt * 16 + lm;

    const unsigned short* wrow = Wo   + (size_t)row * MID;
    const unsigned short* arow = actT + ((size_t)b * SPAT + scol) * MID;

    v8f acc;
    #pragma unroll
    for (int i = 0; i < 8; ++i) acc[i] = 0.f;

    #pragma unroll
    for (int kk = 0; kk < MID; kk += 32) {
        FragBF16 A, B;
        A.q[0] = *(const uint4*)(wrow + kk + 8 * lg);
        A.q[1] = *(const uint4*)(wrow + kk + 16 + 8 * lg);
        B.q[0] = *(const uint4*)(arow + kk + 8 * lg);
        B.q[1] = *(const uint4*)(arow + kk + 16 + 8 * lg);
        acc = __builtin_amdgcn_wmma_f32_16x16x32_bf16(
                  false, A.v, false, B.v, (short)0, acc, false, false);
    }

    const float* xb = x   + (size_t)b * OUTC * SPAT;
    float*       ob = out + (size_t)b * OUTC * SPAT;
    #pragma unroll
    for (int i = 0; i < 8; ++i) {
        const int o = mt * 16 + i + 8 * lg;
        const float v = leaky(acc[i] + bo[o]);
        ob[(size_t)o * SPAT + scol] = v + xb[(size_t)o * SPAT + scol];
    }
}

// ============================================================================
// launch
// ============================================================================
extern "C" void kernel_launch(void* const* d_in, const int* in_sizes, int n_in,
                              void* d_out, int out_size, void* d_ws, size_t ws_size,
                              hipStream_t stream) {
    const float* x       = (const float*)d_in[0];
    const float* conv1_w = (const float*)d_in[1];
    const float* conv1_b = (const float*)d_in[2];
    const float* conv2_w = (const float*)d_in[3];
    const float* conv2_b = (const float*)d_in[4];
    const float* conv3_w = (const float*)d_in[5];
    const float* conv3_b = (const float*)d_in[6];
    const float* convp_w = (const float*)d_in[7];
    const float* convp_b = (const float*)d_in[8];
    const float* bna_g   = (const float*)d_in[9];
    const float* bna_b   = (const float*)d_in[10];
    const float* bna_m   = (const float*)d_in[11];
    const float* bna_v   = (const float*)d_in[12];
    const float* w1      = (const float*)d_in[13];
    const float* bnb_g   = (const float*)d_in[14];
    const float* bnb_b   = (const float*)d_in[15];
    const float* bnb_m   = (const float*)d_in[16];
    const float* bnb_v   = (const float*)d_in[17];
    const float* w2      = (const float*)d_in[18];
    const float* w2_b    = (const float*)d_in[19];
    const float* convo_w = (const float*)d_in[20];
    const float* convo_b = (const float*)d_in[21];
    float* out = (float*)d_out;

    // workspace layout (256B-aligned sections):
    //   Y    : 8*320*3136*4  = 32,112,640  (x1|x2|x3, f32, (b,ch,s))
    //   att  : 8*32*9*3136*4 = 28,901,376  (softmax weights, f32)
    //   xT   : 8*3136*256*2  = 12,845,056  (bf16, (b,s,c))
    //   actT : 8*3136*256*2  = 12,845,056  (bf16, (b,s,c))
    //   Wf   : 320*256*2     = 163,840     (fused qkv weights, bf16)
    //   Wo   : 256*256*2     = 131,072     (convo weights, bf16)
    //   biasf: 320*4         = 1,280
    char* ws = (char*)d_ws;
    size_t off = 0;
    float* Y = (float*)(ws + off);            off += (size_t)BATCH * YCH * SPAT * 4;
    float* att = (float*)(ws + off);          off += (size_t)BATCH * GRP * K2 * SPAT * 4;
    unsigned short* xT = (unsigned short*)(ws + off);   off += (size_t)BATCH * SPAT * CIN * 2;
    unsigned short* actT = (unsigned short*)(ws + off); off += (size_t)BATCH * SPAT * MID * 2;
    unsigned short* Wf = (unsigned short*)(ws + off);   off += (size_t)YCH * CIN * 2;
    unsigned short* Wo = (unsigned short*)(ws + off);   off += (size_t)OUTC * MID * 2;
    float* biasf = (float*)(ws + off);        off += (size_t)YCH * 4;

    // P1: pack weights (320*256 + 256*256 + 320 threads)
    {
        const int n = YCH * CIN + OUTC * MID + YCH;
        k_prep<<<(n + 255) / 256, 256, 0, stream>>>(
            conv1_w, conv1_b, conv2_w, conv2_b, conv3_w, conv3_b, convo_w,
            Wf, biasf, Wo);
    }
    // P2: transpose x -> xT (bf16): 8 * 8 * 98 = 6272 blocks
    k_transpose_x<<<BATCH * (CIN / 32) * (SPAT / 32), 256, 0, stream>>>(x, xT);

    // K1: 8 * 20 * 196 = 31360 waves, 8 waves/block
    k_qkv_gemm<<<(BATCH * (YCH / 16) * (SPAT / 16)) / 8, 256, 0, stream>>>(
        Wf, biasf, xT, Y);

    // K2: 8 * 9 * 3136 = 225,792 threads
    {
        const int n = BATCH * K2 * SPAT;
        k_logits<<<(n + 255) / 256, 256, 0, stream>>>(
            Y, convp_w, convp_b, bna_g, bna_b, bna_m, bna_v, w1,
            bnb_g, bnb_b, bnb_m, bnb_v, w2, w2_b, att);
    }
    // K3: 8 * 32 * 3136 = 802,816 threads
    {
        const int n = BATCH * GRP * SPAT;
        k_softmax<<<(n + 255) / 256, 256, 0, stream>>>(att);
    }
    // K4: 8 * 32 * 3136 threads (each handles a share-group of 8 channels)
    {
        const int n = BATCH * GRP * SPAT;
        k_aggregate<<<(n + 255) / 256, 256, 0, stream>>>(Y, att, actT);
    }
    // K5: 8 * 16 * 196 = 25088 waves, 8 waves/block
    k_out_gemm<<<(BATCH * (OUTC / 16) * (SPAT / 16)) / 8, 256, 0, stream>>>(
        actT, Wo, convo_b, x, out);
}
